// _VectorToTokens_58995670778113
// MI455X (gfx1250) — matrix operand order repaired and checked by reading, not compiled, not run on hardware
//
#include <hip/hip_runtime.h>
#include <stdint.h>

#define BATCH 64
#define FDIM 20000
#define KSEL 2048
#define DMODEL 256
#define HIDDEN 128
#define COORD_SCALE 1e-6f
#define LN_EPS 1e-5f

typedef __attribute__((ext_vector_type(16))) _Float16 v16h;
typedef __attribute__((ext_vector_type(8)))  float    v8f;

// ---------------------------------------------------------------------------
// Kernel 1: per-row top-K (sorted descending, ties -> smaller index first)
// Non-negative floats => IEEE bit pattern is order-preserving.
// Histogram-select threshold bin, compact candidates, bitonic-sort 4096.
// ---------------------------------------------------------------------------
#define TK_THREADS 512
#define CAND 4096

__global__ __launch_bounds__(TK_THREADS)
void topk_kernel(const float* __restrict__ x, float* __restrict__ valsOut,
                 int* __restrict__ idxOut) {
  __shared__ unsigned hist[2048];
  __shared__ unsigned long long cand[CAND];
  __shared__ unsigned s_cnt;
  __shared__ int s_T;

  const int b = blockIdx.x;
  const int tid = threadIdx.x;
  const float* row = x + (size_t)b * FDIM;

  for (int i = tid; i < 2048; i += TK_THREADS) hist[i] = 0u;
  if (tid == 0) s_cnt = 0u;
  __syncthreads();

  for (int f = tid; f < FDIM; f += TK_THREADS) {
    unsigned key = __float_as_uint(__builtin_nontemporal_load(row + f));
    atomicAdd(&hist[key >> 21], 1u);
  }
  __syncthreads();

  if (tid == 0) {
    int total = 0, T = 0;
    for (int bin = 2047; bin >= 0; --bin) {
      total += (int)hist[bin];
      if (total >= KSEL) { T = bin; break; }
    }
    s_T = T;
  }
  __syncthreads();
  const int T = s_T;

  for (int f = tid; f < FDIM; f += TK_THREADS) {
    unsigned key = __float_as_uint(row[f]);   // L2-hot from first pass
    if ((int)(key >> 21) >= T) {
      unsigned p = atomicAdd(&s_cnt, 1u);
      if (p < CAND)
        cand[p] = ((unsigned long long)key << 32) |
                  (unsigned long long)(0xFFFFFFFFu - (unsigned)f);
    }
  }
  __syncthreads();
  unsigned nc = s_cnt; if (nc > CAND) nc = CAND;
  for (unsigned i = nc + tid; i < CAND; i += TK_THREADS) cand[i] = 0ull;

  // bitonic sort, final order descending
  for (unsigned k2 = 2; k2 <= CAND; k2 <<= 1) {
    for (unsigned j = k2 >> 1; j > 0; j >>= 1) {
      __syncthreads();
      for (unsigned i = tid; i < CAND; i += TK_THREADS) {
        unsigned ixj = i ^ j;
        if (ixj > i) {
          unsigned long long a = cand[i], c = cand[ixj];
          bool desc = ((i & k2) == 0);
          if (desc ? (a < c) : (a > c)) { cand[i] = c; cand[ixj] = a; }
        }
      }
    }
  }
  __syncthreads();

  for (int t = tid; t < KSEL; t += TK_THREADS) {
    unsigned long long kv = cand[t];
    valsOut[(size_t)b * KSEL + t] = __uint_as_float((unsigned)(kv >> 32));
    idxOut[(size_t)b * KSEL + t]  = (int)(0xFFFFFFFFu - (unsigned)(kv & 0xFFFFFFFFull));
  }
}

// ---------------------------------------------------------------------------
// Kernel 2: pre-swizzle stacked W2 = [vp_w2 ; cm_w2] (256K x 256N) into f16
// WMMA B-fragment layout: per fragment (kstep, ntile) 32 lanes x 16 halves.
// Lane L: N = ntile*16 + (L&15); K = kstep*32 + (L>=16 ? 16:0) + e, e=0..15.
// ---------------------------------------------------------------------------
__global__ void bfrag_kernel(const float* __restrict__ vp_w2,
                             const float* __restrict__ cm_w2,
                             _Float16* __restrict__ Bfrag) {
  int gid = blockIdx.x * blockDim.x + threadIdx.x;
  if (gid >= 128 * 32) return;           // 8 ksteps * 16 ntiles * 32 lanes
  int fragId = gid >> 5;
  int lane   = gid & 31;
  int ks     = fragId >> 4;
  int ntile  = fragId & 15;
  int n  = ntile * 16 + (lane & 15);
  int kb = ks * 32 + ((lane >= 16) ? 16 : 0);
  _Float16* dst = Bfrag + (size_t)gid * 16;
#pragma unroll
  for (int e = 0; e < 16; ++e) {
    int k = kb + e;
    float w = (k < HIDDEN) ? vp_w2[(size_t)k * DMODEL + n]
                           : cm_w2[(size_t)(k - HIDDEN) * DMODEL + n];
    dst[e] = (_Float16)w;
  }
}

// ---------------------------------------------------------------------------
// CDNA5 async global->LDS copy (ASYNCcnt-tracked, no VGPR round-trip).
// INST_OFFSET is added to both the global and the LDS address (ISA 15.18.3).
// ---------------------------------------------------------------------------
__device__ __forceinline__ void async_g2l_b128(unsigned lds_byte, const void* g) {
  asm volatile("global_load_async_to_lds_b128 %0, %1, off"
               :: "v"(lds_byte), "v"(g) : "memory");
}
__device__ __forceinline__ void async_g2l_b128_off16(unsigned lds_byte, const void* g) {
  asm volatile("global_load_async_to_lds_b128 %0, %1, off offset:16"
               :: "v"(lds_byte), "v"(g) : "memory");
}
__device__ __forceinline__ void wait_asynccnt0() {
#if __has_builtin(__builtin_amdgcn_s_wait_asynccnt)
  __builtin_amdgcn_s_wait_asynccnt(0);
#else
  asm volatile("s_wait_asynccnt 0" ::: "memory");
#endif
}

// ---------------------------------------------------------------------------
// Kernel 3: fused token kernel. 64 tokens / block, 8 waves.
//  step 1: LN -> Linear(3|2 -> 128) -> GELU, both paths, f16 A-tiles in LDS
//  step 2: (64 x 256K) x (256K x 256N) GEMM via v_wmma_f32_16x16x32_f16,
//          B staged per K-step with global_load_async_to_lds_b128
//  step 3: + biases + id_emb gather + chrom_emb gather -> out (NT stores)
// ---------------------------------------------------------------------------
#define MT   64
#define ASTR 264   // halves per A row (padded: banks spread by 4 per M)
#define BSTR 24    // halves per B fragment-lane entry (48B stride)

__global__ __launch_bounds__(256)
void token_kernel(const float* __restrict__ vals, const int* __restrict__ idxb,
                  const _Float16* __restrict__ Bfrag,
                  const float* __restrict__ vp_g, const float* __restrict__ vp_b,
                  const float* __restrict__ vp_w1, const float* __restrict__ vp_b1,
                  const float* __restrict__ cm_g, const float* __restrict__ cm_b,
                  const float* __restrict__ cm_w1, const float* __restrict__ cm_b1,
                  const float* __restrict__ vp_b2, const float* __restrict__ cm_b2,
                  const float* __restrict__ id_emb, const float* __restrict__ chrom_emb,
                  const int* __restrict__ chrom_ids,
                  const float* __restrict__ startp, const float* __restrict__ endp,
                  float* __restrict__ out) {
  __shared__ _Float16 Ash[MT * ASTR];      // 33792 B
  __shared__ _Float16 Bsh[512 * BSTR];     // 24576 B

  const int b     = blockIdx.x >> 5;
  const int kTile = blockIdx.x & 31;
  const int tid   = threadIdx.x;

  // ---- step 1: hidden activations (stacked 256-wide) ----
  {
    const int t     = tid >> 2;           // token 0..63
    const int chunk = tid & 3;            // 64-channel chunk
    const int kk    = kTile * MT + t;
    const float v   = vals[(size_t)b * KSEL + kk];
    const int fidx  = idxb[(size_t)b * KSEL + kk];
    const float r   = (float)kk * (1.0f / (float)(KSEL - 1));
    const float dr  = (v == 0.0f) ? 1.0f : 0.0f;

    // LayerNorm over 3 (val path)
    float mu = (v + r + dr) * (1.0f / 3.0f);
    float a0 = v - mu, a1 = r - mu, a2 = dr - mu;
    float var = (a0 * a0 + a1 * a1 + a2 * a2) * (1.0f / 3.0f);
    float rs = rsqrtf(var + LN_EPS);
    float x0 = a0 * rs * vp_g[0] + vp_b[0];
    float x1 = a1 * rs * vp_g[1] + vp_b[1];
    float x2 = a2 * rs * vp_g[2] + vp_b[2];

    // LayerNorm over 2 (coord path)
    float p0 = startp[fidx] * COORD_SCALE;
    float p1 = endp[fidx]   * COORD_SCALE;
    float cmu = 0.5f * (p0 + p1);
    float c0 = p0 - cmu, c1 = p1 - cmu;
    float cvar = 0.5f * (c0 * c0 + c1 * c1);
    float crs = rsqrtf(cvar + LN_EPS);
    float y0 = c0 * crs * cm_g[0] + cm_b[0];
    float y1 = c1 * crs * cm_g[1] + cm_b[1];

    for (int cc = 0; cc < 64; ++cc) {
      int ch = chunk * 64 + cc;
      float h;
      if (ch < HIDDEN) {
        h = vp_b1[ch] + x0 * vp_w1[ch] + x1 * vp_w1[HIDDEN + ch] + x2 * vp_w1[2 * HIDDEN + ch];
      } else {
        int j = ch - HIDDEN;
        h = cm_b1[j] + y0 * cm_w1[j] + y1 * cm_w1[HIDDEN + j];
      }
      float g = 0.5f * h * (1.0f + erff(h * 0.70710678118654752f)); // exact GELU
      Ash[t * ASTR + ch] = (_Float16)g;
    }
  }

  const int wave   = tid >> 5;
  const int lane   = tid & 31;
  const int mStrip = wave & 3;   // 4 strips of 16 tokens
  const int nHalf  = wave >> 2;  // 2 halves of N=256

  v8f acc[8] = {};
  const int mRow = mStrip * 16 + (lane & 15);
  const unsigned bshBase = (unsigned)(size_t)(&Bsh[0]);

  for (int ks = 0; ks < 8; ++ks) {
    __syncthreads();   // prior-iter consumers done (also orders Ash writes)
    // stage all 16 B fragments of this k-step (16 KB) via async DMA to LDS
    for (int e = tid; e < 512; e += 256) {
      const void* g = (const void*)(Bfrag + ((size_t)ks * 512 + e) * 16);
      unsigned l = bshBase + (unsigned)e * (BSTR * 2);
      async_g2l_b128(l, g);        // bytes [0,16)  of entry
      async_g2l_b128_off16(l, g);  // bytes [16,32) of entry
    }
    if (ks < 7)
      __builtin_prefetch(Bfrag + ((size_t)(ks + 1) * 512) * 16, 0, 1);
    wait_asynccnt0();
    __syncthreads();

    // A fragment: 16x32 f16, contiguous k = 16h + s (same order as Bfrag)
    union AF { v16h v; unsigned u[8]; } af;
#pragma unroll
    for (int vv = 0; vv < 8; ++vv) {
      int k0 = ks * 32 + ((lane >= 16) ? 16 : 0) + 2 * vv;
      af.u[vv] = *(const unsigned*)(Ash + mRow * ASTR + k0);
    }

    // B fragments from LDS; compiler-tracked ds loads + waits. The previous
    // WMMA executes in the matrix pipe while the next pair of ds_load_b128
    // completes, and the second wave per SIMD covers the remainder.
#pragma unroll
    for (int nt = 0; nt < 8; ++nt) {
      union BF { v16h v; uint4 q[2]; } bf;
      const uint4* p = (const uint4*)(Bsh + ((nHalf * 8 + nt) * 32 + lane) * BSTR);
      bf.q[0] = p[0];
      bf.q[1] = p[1];
      acc[nt] = __builtin_amdgcn_wmma_f32_16x16x32_f16(
          false, af.v, false, bf.v, (short)0, acc[nt], false, false);
    }
  }

  // ---- step 3: epilogue with fused biases + gathers; NT stores ----
  const int rowBase = kTile * MT;
#pragma unroll
  for (int r = 0; r < 8; ++r) {
    int m   = mStrip * 16 + ((lane < 16) ? r : 8 + r);
    int tok = rowBase + m;
    int fidx = idxb[(size_t)b * KSEL + tok];
    int cid  = chrom_ids[fidx];
    const float* ide = id_emb    + (size_t)fidx * DMODEL;
    const float* che = chrom_emb + (size_t)cid  * DMODEL;
    float* orow = out + ((size_t)b * (KSEL + 1) + 1 + tok) * DMODEL;
#pragma unroll
    for (int nt = 0; nt < 8; ++nt) {
      int n = (nHalf * 8 + nt) * 16 + (lane & 15);
      float val = acc[nt][r] + vp_b2[n] + cm_b2[n] + ide[n] + che[n];
      __builtin_nontemporal_store(val, orow + n);
    }
  }
}

// ---------------------------------------------------------------------------
// Kernel 4: CLS row
// ---------------------------------------------------------------------------
__global__ void cls_kernel(const float* __restrict__ cls_tok,
                           float* __restrict__ out) {
  int b = blockIdx.x;
  int ch = threadIdx.x;
  __builtin_nontemporal_store(cls_tok[ch],
                              out + (size_t)b * (KSEL + 1) * DMODEL + ch);
}

// ---------------------------------------------------------------------------
extern "C" void kernel_launch(void* const* d_in, const int* in_sizes, int n_in,
                              void* d_out, int out_size, void* d_ws, size_t ws_size,
                              hipStream_t stream) {
  (void)in_sizes; (void)n_in; (void)out_size; (void)ws_size;
  const float* x        = (const float*)d_in[0];
  const float* vp_g     = (const float*)d_in[1];
  const float* vp_b     = (const float*)d_in[2];
  const float* vp_w1    = (const float*)d_in[3];
  const float* vp_b1    = (const float*)d_in[4];
  const float* vp_w2    = (const float*)d_in[5];
  const float* vp_b2    = (const float*)d_in[6];
  const float* id_emb   = (const float*)d_in[7];
  const float* chrom_emb= (const float*)d_in[8];
  const float* cm_g     = (const float*)d_in[9];
  const float* cm_b     = (const float*)d_in[10];
  const float* cm_w1    = (const float*)d_in[11];
  const float* cm_b1    = (const float*)d_in[12];
  const float* cm_w2    = (const float*)d_in[13];
  const float* cm_b2    = (const float*)d_in[14];
  const float* cls_tok  = (const float*)d_in[15];
  const int*   chrom_ids= (const int*)d_in[16];
  const float* startp   = (const float*)d_in[17];
  const float* endp     = (const float*)d_in[18];
  float* out = (float*)d_out;

  char* ws = (char*)d_ws;
  float*    valsWS = (float*)ws;                     // 64*2048*4 = 512 KB
  int*      idxWS  = (int*)(ws + 512 * 1024);        // 512 KB
  _Float16* Bfrag  = (_Float16*)(ws + 1024 * 1024);  // 128 KB

  topk_kernel<<<BATCH, TK_THREADS, 0, stream>>>(x, valsWS, idxWS);
  bfrag_kernel<<<16, 256, 0, stream>>>(vp_w2, cm_w2, Bfrag);
  token_kernel<<<BATCH * 32, 256, 0, stream>>>(
      valsWS, idxWS, Bfrag, vp_g, vp_b, vp_w1, vp_b1,
      cm_g, cm_b, cm_w1, cm_b1, vp_b2, cm_b2,
      id_emb, chrom_emb, chrom_ids, startp, endp, out);
  cls_kernel<<<BATCH, DMODEL, 0, stream>>>(cls_tok, out);
}
